// UpdateGrid_27401891348991
// MI455X (gfx1250) — matrix operand order repaired and checked by reading, not compiled, not run on hardware
//
#include <hip/hip_runtime.h>

typedef __attribute__((ext_vector_type(16))) _Float16 v16h;
typedef __attribute__((ext_vector_type(8)))  float    v8f;
typedef __attribute__((ext_vector_type(2)))  float    v2f;

namespace {
constexpr int kB = 32, kC = 16, kH = 256, kW = 256, kHID = 128;
constexpr int kHW = kH * kW;            // 65536
constexpr int kPix = kB * kHW;          // 2097152
constexpr float kFltMax = 3.402823466e38f;
}

// Zero-padded tap into a [B,H,W] float plane ('SAME' conv padding).
__device__ __forceinline__ float tap(const float* __restrict__ buf, int b, int y, int x) {
  return ((unsigned)y < (unsigned)kH && (unsigned)x < (unsigned)kW)
             ? buf[((size_t)b * kH + y) * kW + x] : 0.f;
}

// K0: w1eff[k] = {sum(w1[k,0:16]), sum(w1[k,16:32]), sum(w1[k,32:48]), b1[k]}
// (slot 3 = b1 so the h-GEMM can fold bias via the K=3 column of A with B[K=3]=1.0)
__global__ void nca_prep(const float* __restrict__ w1, const float* __restrict__ b1,
                         float* __restrict__ w1eff) {
  int k = threadIdx.x;
  if (k < kHID) {
    float s0 = 0.f, s1 = 0.f, s2 = 0.f;
    #pragma unroll
    for (int c = 0; c < 16; ++c) {
      s0 += w1[k * 48 + c];
      s1 += w1[k * 48 + 16 + c];
      s2 += w1[k * 48 + 32 + c];
    }
    w1eff[4 * k + 0] = s0;
    w1eff[4 * k + 1] = s1;
    w1eff[4 * k + 2] = s2;
    w1eff[4 * k + 3] = b1[k];
  }
}

// K1: xs[b,y,x] = sum_c x[b,c,y,x]
__global__ void nca_chansum(const float* __restrict__ x, float* __restrict__ xs) {
  int idx = blockIdx.x * blockDim.x + threadIdx.x;   // 0..kPix-1 (exact)
  int b = idx / kHW;
  int pix = idx - b * kHW;
  const float* px = x + (size_t)b * kC * kHW + pix;
  float s = 0.f;
  #pragma unroll
  for (int c = 0; c < kC; ++c) s += px[(size_t)c * kHW];
  xs[idx] = s;
}

// K2: alpha flag = ((x[b,3] + (w2[3,:]·relu(W1eff·p+b1) + b2[3])*mask) > 0.1) ? 1 : 0
__global__ void nca_alpha(const float* __restrict__ x, const int* __restrict__ mask,
                          const float* __restrict__ w2, const float* __restrict__ b2,
                          const float* __restrict__ w1eff, const float* __restrict__ xs,
                          float* __restrict__ alpha) {
  __shared__ float lw1[kHID * 4];
  __shared__ float lw2r3[kHID];
  for (int i = threadIdx.x; i < kHID * 4; i += blockDim.x) lw1[i] = w1eff[i];
  for (int i = threadIdx.x; i < kHID; i += blockDim.x) lw2r3[i] = w2[3 * kHID + i];
  __syncthreads();

  int idx = blockIdx.x * blockDim.x + threadIdx.x;
  int b = idx / kHW;
  int pix = idx - b * kHW;
  int y = pix / kW, xx = pix - y * kW;

  float n00 = tap(xs, b, y - 1, xx - 1), n01 = tap(xs, b, y - 1, xx), n02 = tap(xs, b, y - 1, xx + 1);
  float n10 = tap(xs, b, y,     xx - 1), n11 = tap(xs, b, y,     xx), n12 = tap(xs, b, y,     xx + 1);
  float n20 = tap(xs, b, y + 1, xx - 1), n21 = tap(xs, b, y + 1, xx), n22 = tap(xs, b, y + 1, xx + 1);
  float gx = (n02 - n00) + 2.f * (n12 - n10) + (n22 - n20);
  float gy = (n20 + 2.f * n21 + n22) - (n00 + 2.f * n01 + n02);
  float pid = n11;

  float acc = 0.f;
  #pragma unroll 8
  for (int k = 0; k < kHID; ++k) {
    float hv = fmaf(lw1[4 * k], gx, fmaf(lw1[4 * k + 1], gy, fmaf(lw1[4 * k + 2], pid, lw1[4 * k + 3])));
    hv = __builtin_amdgcn_fmed3f(hv, 0.f, kFltMax);   // relu, single v_med3
    acc = fmaf(lw2r3[k], hv, acc);
  }
  float mk = (float)mask[idx];
  float x3 = x[((size_t)b * kC + 3) * kHW + pix];
  float a3 = fmaf(acc + b2[3], mk, x3);
  alpha[idx] = (a3 > 0.1f) ? 1.f : 0.f;
}

// K3: h via v_wmma_f32_16x16x4_f32 (8 tiles of 16 hidden rows, K=4 with K3 bias trick),
//     dx via v_wmma_f32_16x16x32_f16 (M=16 out-ch, N=16 pixels, K=128 in 4 chunks).
//     The h-GEMM A-rows are PERMUTED (hidden = 32ci + 8j + m + (m>=8 ? 8 : 0)) so the
//     f32-WMMA D layout lands exactly in the f16-WMMA B layout — no cross-lane shuffles.
//     1 wave = 4 pixel-tiles = 64 consecutive pixels; all weight fragments hoisted.
__global__ void __launch_bounds__(256) nca_update(
    const float* __restrict__ x, const int* __restrict__ mask,
    const float* __restrict__ w2, const float* __restrict__ b2,
    const float* __restrict__ w1eff, const float* __restrict__ xs,
    const float* __restrict__ alpha, float* __restrict__ out) {
  int lane = threadIdx.x & 31;
  int p    = lane & 15;          // pixel within tile == A-row M of dx-GEMM == B-col N
  int hi   = lane >> 4;
  int wg   = blockIdx.x * 8 + (threadIdx.x >> 5);   // wave-group id, 32768 total
  int b    = wg >> 10;                               // 1024 wave-groups per batch image
  int rem  = wg & 1023;
  int y    = rem >> 2;
  int xbase = (rem & 3) << 6;                        // 64-pixel row segment

  // --- hoisted A fragments for the h-GEMM (16x4 f32 layout, permuted rows) ---
  // lane<16 holds {col0, col1}; lane>=16 holds {col2, b1}: w1eff[4*hrow + 2*hi + j]
  v2f w1A[8];
  #pragma unroll
  for (int ci = 0; ci < 4; ++ci) {
    #pragma unroll
    for (int j = 0; j < 2; ++j) {
      int hrow = 32 * ci + 8 * j + p + ((p >= 8) ? 8 : 0);   // row permutation
      const float* src = w1eff + 4 * hrow + 2 * hi;
      v2f a; a[0] = src[0]; a[1] = src[1];
      w1A[2 * ci + j] = a;
    }
  }

  // --- hoisted A fragments for the dx-GEMM: w2 rows (f32 -> f16), contiguous k = 32ci + 16*hi + e order ---
  v16h w2A[4];
  #pragma unroll
  for (int ci = 0; ci < 4; ++ci) {
    #pragma unroll
    for (int e = 0; e < 16; ++e) {
      int k = 32 * ci + e + hi * 16;
      w2A[ci][e] = (_Float16)w2[p * kHID + k];
    }
  }

  float bias8[8];
  #pragma unroll
  for (int r = 0; r < 8; ++r) bias8[r] = b2[hi * 8 + r];

  v8f zero8;
  #pragma unroll
  for (int r = 0; r < 8; ++r) zero8[r] = 0.f;

  #pragma unroll
  for (int ti = 0; ti < 4; ++ti) {
    int xx = xbase + ti * 16 + p;

    // --- perception (3x3 over xs, zero-padded) ---
    float n00 = tap(xs, b, y - 1, xx - 1), n01 = tap(xs, b, y - 1, xx), n02 = tap(xs, b, y - 1, xx + 1);
    float n10 = tap(xs, b, y,     xx - 1), n11 = tap(xs, b, y,     xx), n12 = tap(xs, b, y,     xx + 1);
    float n20 = tap(xs, b, y + 1, xx - 1), n21 = tap(xs, b, y + 1, xx), n22 = tap(xs, b, y + 1, xx + 1);
    float gx = (n02 - n00) + 2.f * (n12 - n10) + (n22 - n20);
    float gy = (n20 + 2.f * n21 + n22) - (n00 + 2.f * n01 + n02);
    float pid = n11;

    // B fragment of h-GEMM (4x16 f32): lanes<16 hold {K0,K1}={gx,gy}, lanes>=16 {K2,K3}={pid,1}
    v2f bp;
    bp[0] = hi ? pid : gx;
    bp[1] = hi ? 1.0f : gy;

    v8f acc;
    #pragma unroll
    for (int r = 0; r < 8; ++r) acc[r] = bias8[r];

    #pragma unroll
    for (int ci = 0; ci < 4; ++ci) {
      // h pre-activations for hidden K-range [32ci, 32ci+32), rows permuted to B layout
      v8f d0 = __builtin_amdgcn_wmma_f32_16x16x4_f32(false, w1A[2 * ci],     false, bp,
                                                     (short)0, zero8, false, false);
      v8f d1 = __builtin_amdgcn_wmma_f32_16x16x4_f32(false, w1A[2 * ci + 1], false, bp,
                                                     (short)0, zero8, false, false);
      // relu (v_med3) + pack straight into the f16 B fragment — layouts already match
      v16h bm;
      #pragma unroll
      for (int r = 0; r < 8; ++r) {
        bm[r]     = (_Float16)__builtin_amdgcn_fmed3f(d0[r], 0.f, kFltMax);
        bm[r + 8] = (_Float16)__builtin_amdgcn_fmed3f(d1[r], 0.f, kFltMax);
      }
      acc = __builtin_amdgcn_wmma_f32_16x16x32_f16(false, w2A[ci], false, bm,
                                                   (short)0, acc, false, false);
    }

    // --- alive gate: 3x3 any() over alpha flags (zero-padded) ---
    float s = tap(alpha, b, y - 1, xx - 1) + tap(alpha, b, y - 1, xx) + tap(alpha, b, y - 1, xx + 1)
            + tap(alpha, b, y,     xx - 1) + tap(alpha, b, y,     xx) + tap(alpha, b, y,     xx + 1)
            + tap(alpha, b, y + 1, xx - 1) + tap(alpha, b, y + 1, xx) + tap(alpha, b, y + 1, xx + 1);
    float alivef = (s > 0.f) ? 1.f : 0.f;
    float mk = (float)mask[((size_t)b * kH + y) * kW + xx];

    // D layout: lane p holds channels 0..7, lane p+16 holds channels 8..15 (pixel N=p)
    size_t base = (size_t)b * kC * kHW + (size_t)y * kW + xx;
    #pragma unroll
    for (int r = 0; r < 8; ++r) {
      int c = hi * 8 + r;
      size_t idx = base + (size_t)c * kHW;
      out[idx] = fmaf(acc[r], mk, x[idx]) * alivef;
    }
  }
}

extern "C" void kernel_launch(void* const* d_in, const int* in_sizes, int n_in,
                              void* d_out, int out_size, void* d_ws, size_t ws_size,
                              hipStream_t stream) {
  const float* x    = (const float*)d_in[0];
  const int*   mask = (const int*)d_in[1];
  const float* w1   = (const float*)d_in[2];
  const float* b1   = (const float*)d_in[3];
  const float* w2   = (const float*)d_in[4];
  const float* b2   = (const float*)d_in[5];
  float* out = (float*)d_out;

  // workspace layout: [w1eff 2KB pad to 4KB][xs 8MB][alpha 8MB]
  float* w1eff = (float*)d_ws;
  float* xs    = (float*)((char*)d_ws + 4096);
  float* alpha = (float*)((char*)d_ws + 4096 + (size_t)kPix * sizeof(float));

  nca_prep<<<1, 128, 0, stream>>>(w1, b1, w1eff);
  nca_chansum<<<kPix / 256, 256, 0, stream>>>(x, xs);
  nca_alpha<<<kPix / 256, 256, 0, stream>>>(x, mask, w2, b2, w1eff, xs, alpha);
  // 32768 wave-groups (4 tiles each), 8 waves per 256-thread block
  nca_update<<<32768 / 8, 256, 0, stream>>>(x, mask, w2, b2, w1eff, xs, alpha, out);
}
